// SiameseGNN_41626823033581
// MI455X (gfx1250) — matrix operand order repaired and checked by reading, not compiled, not run on hardware
//
#include <hip/hip_runtime.h>
#include <hip/hip_bf16.h>

#define NN  65536
#define EE  1048576
#define FIN 128
#define FH  64   // HID == OUT == 64

typedef __attribute__((ext_vector_type(16))) __bf16 v16bf;
typedef __attribute__((ext_vector_type(8)))  float  v8f;

// ---------------- degree / normalization ----------------

__global__ __launch_bounds__(256) void k_zero_deg(unsigned* deg) {
    deg[blockIdx.x * 256 + threadIdx.x] = 0u;
}

__global__ __launch_bounds__(256) void k_deg(const int* __restrict__ dst, unsigned* __restrict__ deg) {
    int e = blockIdx.x * 256 + threadIdx.x;
    atomicAdd(&deg[dst[e]], 1u);
}

__global__ __launch_bounds__(256) void k_dinv(const unsigned* __restrict__ deg, float* __restrict__ dinv) {
    int i = blockIdx.x * 256 + threadIdx.x;
    dinv[i] = rsqrtf((float)(deg[i] + 1u));   // +1 = self loop
}

// ---------------- dense transform: H = X @ W  (bf16 WMMA) ----------------
// W is pre-swizzled in LDS into per-lane B fragments:
//   sB[kkblk*4 + nb][lane] = the 16 contiguous bf16 this lane feeds to
//   v_wmma_f32_16x16x32_bf16 for column block nb at K offset kkblk*32.
// B fragment layout (32x16, K x N): lanes 0-15 hold col = lane, K = kk..kk+15;
//                                   lanes 16-31 hold col = lane-16, K = kk+16..kk+31.
// A fragment (16x32): lane l<16 holds row l, K = kk+[0..15] (same k order as B);
//                     lane l>=16 holds row l-16, K = kk+[16..31].
// D (16x16 f32): vgpr v -> row = v + 8*(lane>=16), col = lane&15.
// One block = 8 waves; each wave produces 2 row-tiles (32 rows) x 64 cols,
// reusing each B fragment across both tiles.
template <int K>
__global__ __launch_bounds__(256)
void k_gemm_wmma(const float* __restrict__ X, const float* __restrict__ W, float* __restrict__ H) {
    constexpr int NFRAG = (K / 32) * 4;           // fragments: K-blocks x col-blocks
    __shared__ v16bf sB[NFRAG * 32];              // 32 lanes each, 32 B/lane

    // ---- one-time pack: global W (f32, [K][64]) -> swizzled bf16 fragments ----
    for (int p = threadIdx.x; p < NFRAG * 32; p += 256) {
        const int plane  = p & 31;
        const int nb     = (p >> 5) & 3;
        const int kkblk  = p >> 7;
        const int col    = nb * 16 + (plane & 15);
        const int k0     = kkblk * 32 + (plane >> 4) * 16;
        v16bf tmp;
#pragma unroll
        for (int j = 0; j < 16; ++j) tmp[j] = (__bf16)W[(k0 + j) * FH + col];
        sB[p] = tmp;
    }
    __syncthreads();

    const int wave = threadIdx.x >> 5;
    const int lane = threadIdx.x & 31;
    const int hl   = lane >> 4;          // half-wave selector
    const int l15  = lane & 15;
    const int row0 = (blockIdx.x * 8 + wave) * 32;   // 2 tiles of 16 rows

    v8f acc[2][4] = {};

#pragma unroll
    for (int kk = 0; kk < K; kk += 32) {
        const int kkblk = kk >> 5;

        // ---- B fragments: one vector LDS load each (2x ds_load_b128) ----
        v16bf b0 = sB[(kkblk * 4 + 0) * 32 + lane];
        v16bf b1 = sB[(kkblk * 4 + 1) * 32 + lane];
        v16bf b2 = sB[(kkblk * 4 + 2) * 32 + lane];
        v16bf b3 = sB[(kkblk * 4 + 3) * 32 + lane];

#pragma unroll
        for (int t = 0; t < 2; ++t) {
            // ---- A fragment from global (f32 -> packed bf16 cvt) ----
            const int r  = row0 + t * 16 + l15;
            const int ka = kk + hl * 16;
            float af[16];
            const float4* xp = (const float4*)(X + (size_t)r * K + ka);
            *(float4*)&af[0]  = xp[0];
            *(float4*)&af[4]  = xp[1];
            const float4* xq = (const float4*)(X + (size_t)r * K + ka + 8);
            *(float4*)&af[8]  = xq[0];
            *(float4*)&af[12] = xq[1];
            v16bf a;
#pragma unroll
            for (int j = 0; j < 16; ++j) a[j] = (__bf16)af[j];

            acc[t][0] = __builtin_amdgcn_wmma_f32_16x16x32_bf16(false, a, false, b0, (short)0, acc[t][0], false, false);
            acc[t][1] = __builtin_amdgcn_wmma_f32_16x16x32_bf16(false, a, false, b1, (short)0, acc[t][1], false, false);
            acc[t][2] = __builtin_amdgcn_wmma_f32_16x16x32_bf16(false, a, false, b2, (short)0, acc[t][2], false, false);
            acc[t][3] = __builtin_amdgcn_wmma_f32_16x16x32_bf16(false, a, false, b3, (short)0, acc[t][3], false, false);
        }
    }

    // ---- store D tiles ----
#pragma unroll
    for (int t = 0; t < 2; ++t) {
#pragma unroll
        for (int nb = 0; nb < 4; ++nb) {
            const int col = nb * 16 + l15;
#pragma unroll
            for (int v = 0; v < 8; ++v) {
                H[(size_t)(row0 + t * 16 + v + hl * 8) * FH + col] = acc[t][nb][v];
            }
        }
    }
}

// ---------------- aggregation ----------------
// agg[i,f] = h[i,f] * dinv[i]^2 + bias[f]   (self-loop term + bias folded in)
__global__ __launch_bounds__(256)
void k_selfinit(const float* __restrict__ h, const float* __restrict__ dinv,
                const float* __restrict__ bias, float* __restrict__ agg) {
    int idx = blockIdx.x * 256 + threadIdx.x;
    int i = idx >> 6, f = idx & 63;
    float di = dinv[i];
    agg[idx] = h[idx] * di * di + bias[f];
}

// agg[dst,f] += h[src,f] * dinv[src]*dinv[dst]  via hardware f32 atomics in L2
__global__ __launch_bounds__(256)
void k_scatter(const float* __restrict__ h, const int* __restrict__ src,
               const int* __restrict__ dst, const float* __restrict__ dinv,
               float* __restrict__ agg) {
    unsigned idx = blockIdx.x * 256u + threadIdx.x;
    unsigned e = idx >> 6;
    unsigned f = idx & 63u;
    int s = src[e];
    int d = dst[e];
    float norm = dinv[s] * dinv[d];
    float v = h[(size_t)s * FH + f] * norm;
    __hip_atomic_fetch_add(&agg[(size_t)d * FH + f], v,
                           __ATOMIC_RELAXED, __HIP_MEMORY_SCOPE_AGENT);
}

__global__ __launch_bounds__(256)
void k_relu(const float* __restrict__ agg, float* __restrict__ h) {
    int idx = blockIdx.x * 256 + threadIdx.x;
    h[idx] = fmaxf(agg[idx], 0.0f);
}

// ---------------- driver ----------------

extern "C" void kernel_launch(void* const* d_in, const int* in_sizes, int n_in,
                              void* d_out, int out_size, void* d_ws, size_t ws_size,
                              hipStream_t stream) {
    const float* x   = (const float*)d_in[0];
    const int*   ei  = (const int*)d_in[1];
    const float* W1  = (const float*)d_in[2];
    const float* b1  = (const float*)d_in[3];
    const float* W2  = (const float*)d_in[4];
    const float* b2  = (const float*)d_in[5];
    float*       out = (float*)d_out;

    const int* src = ei;        // edge_index[0]
    const int* dst = ei + EE;   // edge_index[1]

    float*    dinv = (float*)d_ws;
    unsigned* deg  = (unsigned*)d_ws + NN;
    float*    bufA = (float*)d_ws + 2 * NN;            // h_t / h    (N x 64)
    float*    bufB = bufA + (size_t)NN * FH;           // agg / h2_t (N x 64)

    // normalization
    k_zero_deg<<<NN / 256, 256, 0, stream>>>(deg);
    k_deg<<<EE / 256, 256, 0, stream>>>(dst, deg);
    k_dinv<<<NN / 256, 256, 0, stream>>>(deg, dinv);

    // layer 1: transform -> aggregate -> relu
    k_gemm_wmma<FIN><<<NN / 256, 256, 0, stream>>>(x, W1, bufA);
    k_selfinit<<<NN * FH / 256, 256, 0, stream>>>(bufA, dinv, b1, bufB);
    k_scatter<<<EE * FH / 256, 256, 0, stream>>>(bufA, src, dst, dinv, bufB);
    k_relu<<<NN * FH / 256, 256, 0, stream>>>(bufB, bufA);

    // layer 2: transform -> aggregate (bias folded into init), result in d_out
    k_gemm_wmma<FH><<<NN / 256, 256, 0, stream>>>(bufA, W2, bufB);
    k_selfinit<<<NN * FH / 256, 256, 0, stream>>>(bufB, dinv, b2, out);
    k_scatter<<<EE * FH / 256, 256, 0, stream>>>(bufB, src, dst, dinv, out);
}
